// PointCloud3DCNN_73040213836401
// MI455X (gfx1250) — matrix operand order repaired and checked by reading, not compiled, not run on hardware
//
#include <hip/hip_runtime.h>
#include <stdint.h>
#include <stddef.h>

// ---------------------------------------------------------------------------
// Types for CDNA5 WMMA (wave32): bf16 A/B fragments, f32 accumulator.
// ---------------------------------------------------------------------------
typedef __bf16 bf16_t;
typedef __attribute__((ext_vector_type(16))) __bf16 v16bf;
typedef __attribute__((ext_vector_type(8)))  float  v8f;
typedef __attribute__((ext_vector_type(4)))  unsigned int v4u;

union ABFrag { v16bf v; v4u u[2]; };

__device__ __forceinline__ bf16_t f2bf(float f) {
  union { float f; unsigned u; } x; x.f = f;
  unsigned r = x.u + 0x7fffu + ((x.u >> 16) & 1u);   // round-to-nearest-even
  union { unsigned short s; bf16_t h; } o; o.s = (unsigned short)(r >> 16);
  return o.h;
}
__device__ __forceinline__ float bf2f(bf16_t v) {
  union { bf16_t h; unsigned short s; } i; i.h = v;
  union { unsigned u; float f; } o; o.u = ((unsigned)i.s) << 16;
  return o.f;
}

// ---------------------------------------------------------------------------
// Weight repack: f32 OIDHWT (Co,Ci,3,3,3,kt) -> bf16 [tap][co][ci_pad],
// tap = sp*kt + kti, sp = (kd*3+kh)*3+kw. grid.y = sp, grid.z = kti -> no
// integer div/mod. Zero-fill padded input channels.
// ---------------------------------------------------------------------------
__global__ void wt_pack_kernel(const float* __restrict__ w, bf16_t* __restrict__ out,
                               int Co, int Ci, int CiPadSh, int kt) {
  const int idx = blockIdx.x * blockDim.x + threadIdx.x;      // over Co*CiPad
  const int CiPad = 1 << CiPadSh;
  if (idx >= (Co << CiPadSh)) return;
  const int cip = idx & (CiPad - 1);
  const int co  = idx >> CiPadSh;
  const int sp  = blockIdx.y;
  const int kti = blockIdx.z;
  const int tap = sp * kt + kti;
  float v = 0.f;
  if (cip < Ci)
    v = w[(((long long)co * Ci + cip) * 27 + sp) * kt + kti];
  out[(long long)tap * Co * CiPad + idx] = f2bf(v);
}

// BatchNorm eval-form folding: scale = g*rsqrt(v+eps), shift = b - m*scale
__global__ void bn_prep_kernel(const float* g, const float* b, const float* m,
                               const float* v, float* scale, float* shift, int C) {
  int c = blockIdx.x * blockDim.x + threadIdx.x;
  if (c >= C) return;
  float s = g[c] * rsqrtf(v[c] + 1e-5f);
  scale[c] = s;
  shift[c] = b[c] - m[c] * s;
}

// Pack x: NCDHWT f32 (9 ch) -> voxel-major [vox][32] bf16 (ch 9..31 zero),
// plus level-0 active mask (any channel nonzero).
__global__ void pack_x_kernel(const float* __restrict__ x, bf16_t* __restrict__ x32,
                              float* __restrict__ mask0, int nvox) {
  int v = blockIdx.x * blockDim.x + threadIdx.x;
  if (v >= nvox) return;
  bool any = false;
#pragma unroll
  for (int c = 0; c < 9; ++c) {
    float f = x[(size_t)c * nvox + v];
    any = any || (f != 0.f);
    x32[(size_t)v * 32 + c] = f2bf(f);
  }
#pragma unroll
  for (int c = 9; c < 32; ++c) x32[(size_t)v * 32 + c] = f2bf(0.f);
  mask0[v] = any ? 1.f : 0.f;
}

// Sparse-conv active-output rule: 3x3x3 (spatial) max-pool, stride 2, pad 1,
// per temporal slice. All dims are powers of two -> shift/mask decode.
__global__ void down_mask_kernel(const float* __restrict__ mi, float* __restrict__ mo,
                                 int Di, int Do, int dsh) {
  int idx = blockIdx.x * blockDim.x + threadIdx.x;
  int nvox = Do * Do * Do * 2;
  if (idx >= nvox) return;
  int t = idx & 1;
  int sp = idx >> 1;
  int w = sp & (Do - 1);
  int h = (sp >> dsh) & (Do - 1);
  int d = sp >> (2 * dsh);
  float m = 0.f;
  for (int kd = -1; kd <= 1; ++kd)
    for (int kh = -1; kh <= 1; ++kh)
      for (int kw = -1; kw <= 1; ++kw) {
        int id = 2 * d + kd, ih = 2 * h + kh, iw = 2 * w + kw;
        if (id < 0 || id >= Di || ih < 0 || ih >= Di || iw < 0 || iw >= Di) continue;
        m = fmaxf(m, mi[(((size_t)id * Di + ih) * Di + iw) * 2 + t]);
      }
  mo[idx] = m;
}

// ---------------------------------------------------------------------------
// Implicit-GEMM conv. One wave = 16 voxels x (NT*16) output channels.
// Weights for the block's co-range are identical across all 8 waves, so each
// tap's contiguous weight slice (NT*16 rows x NC*32 ci, bf16) is staged into
// LDS with gfx1250 async global->LDS copies (ASYNCcnt), double-buffered
// across taps, and consumed via ds_load_b128. A-fragments are per-voxel
// global b128 loads reused across NT co-tiles in registers.
//   NT = co-tiles per wave (1,2,4)
//   KT = temporal kernel (1 or 3)
//   NC = K-chunks of 32 input channels (CiPad/32)
// Epilogue fuses BN scale/shift + ReLU + active-voxel mask, stores bf16.
// ---------------------------------------------------------------------------
struct ConvArgs {
  const bf16_t* in;     // [inVox][NC*32]
  const bf16_t* wt;     // [tap][Co][NC*32]
  bf16_t* out;          // [outVox][CoPad]
  const float* scale;   // [Co]
  const float* shift;   // [Co]
  const float* mask;    // [outVox]
  int inDim;            // cubic input spatial dim
  int outDim;           // cubic output spatial dim (power of two)
  int outSh;            // log2(outDim)
  int stride;           // 1 or 2
  int dil;              // 1 (normal) or 2 (lhs-dilated inverse conv)
  int Co, CoPad;
  int numTiles;
};

__device__ __forceinline__ bool map_coord(int o, int k, int stride, int dil,
                                          int n, int& idx) {
  int j = o * stride + k - 1;                  // pad_lo = 1 for every conv here
  if (dil == 2) {
    if (j & 1) return false;
    j >>= 1;
  }
  if ((unsigned)j >= (unsigned)n) return false;
  idx = j;
  return true;
}

template <int NT, int KT, int NC>
__global__ void __launch_bounds__(256) conv_wmma_kernel(ConvArgs a) {
  constexpr int ROWB    = NC * 64;             // bytes per weight row (CiPad bf16)
  constexpr int STAGE_B = NT * 16 * ROWB;      // bytes per tap weight slice
  constexpr int NTAPS   = 27 * KT;
  __shared__ __align__(16) char smem[2 * STAGE_B];

  const int tid  = threadIdx.x;
  const int lane = tid & 31;
  const int wave = tid >> 5;
  int tile = blockIdx.x * 8 + wave;
  if (tile >= a.numTiles) tile = a.numTiles - 1;  // clamp: keep barrier-uniform
  const int co0    = blockIdx.y * (16 * NT);
  const int lanelo = lane & 15;
  const int hs     = lane >> 4;                // lane half-select

  // This lane's output voxel (A-matrix row M = lanelo). Power-of-two decode.
  const int v   = tile * 16 + lanelo;
  const int ot  = v & 1;
  const int sp  = v >> 1;
  const int Wm  = a.outDim - 1;
  const int ow  = sp & Wm;
  const int oh  = (sp >> a.outSh) & Wm;
  const int od  = sp >> (2 * a.outSh);

  const char* inB    = (const char*)a.in;
  const char* wslice = (const char*)a.wt + (size_t)co0 * ROWB;  // this block's rows
  const int tapStrideB = a.Co * ROWB;          // bytes per tap in packed weights
  const unsigned ldsBase = (unsigned)(uintptr_t)&smem[0];       // LDS addr = low 32b

  // Cooperative async copy of one tap's contiguous weight slice into stage s.
  auto stage_load = [&](int tap, int s) {
    const char* src = wslice + (size_t)tap * tapStrideB;
    const unsigned dstBase = ldsBase + (unsigned)(s * STAGE_B);
#pragma unroll
    for (int off = 0; off < STAGE_B; off += 256 * 16) {
      int o = off + tid * 16;
      if (o < STAGE_B) {
        unsigned lds = dstBase + (unsigned)o;
        unsigned long long g = (unsigned long long)(src + o);
        asm volatile("global_load_async_to_lds_b128 %0, %1, off"
                     :: "v"(lds), "v"(g) : "memory");
      }
    }
  };

  const v8f zf = {0.f, 0.f, 0.f, 0.f, 0.f, 0.f, 0.f, 0.f};
  v8f acc[NT];
#pragma unroll
  for (int n = 0; n < NT; ++n) acc[n] = zf;

  stage_load(0, 0);                            // prologue: tap 0 -> stage 0

#pragma unroll 1
  for (int kd = 0; kd < 3; ++kd)
#pragma unroll 1
    for (int kh = 0; kh < 3; ++kh)
#pragma unroll 1
      for (int kw = 0; kw < 3; ++kw) {
        int id = 0, ih = 0, iw = 0;
        const bool svalid =
            map_coord(od, kd, a.stride, a.dil, a.inDim, id) &&
            map_coord(oh, kh, a.stride, a.dil, a.inDim, ih) &&
            map_coord(ow, kw, a.stride, a.dil, a.inDim, iw);
        const int spTap = (kd * 3 + kh) * 3 + kw;
#pragma unroll
        for (int kti = 0; kti < KT; ++kti) {
          const int tap = spTap * KT + kti;    // monotonically 0..NTAPS-1
          // -- pipeline: wait this wave's async ops, sync block, prefetch next
          asm volatile("s_wait_asynccnt 0" ::: "memory");
          __syncthreads();
          if (tap + 1 < NTAPS) stage_load(tap + 1, (tap + 1) & 1);
          const char* bstage = &smem[(tap & 1) * STAGE_B] + lanelo * ROWB + hs * 32;

          int it = (KT == 3) ? (ot + kti - 1) : ot;
          bool valid = svalid && ((unsigned)it < 2u);
          if (!valid) it = 0;
          const unsigned ivox =
              (unsigned)(((id * a.inDim + ih) * a.inDim + iw) * 2 + it);
          const char* inrow = inB + (size_t)ivox * ROWB;
#pragma unroll
          for (int c = 0; c < NC; ++c) {
            ABFrag A;
            const v4u z = {0u, 0u, 0u, 0u};
            if (valid) {
              // A k order used here (contiguous, matches B): lanes 0-15 -> K 0..15; lanes 16-31 -> K 16..31
              A.u[0] = *(const v4u*)(inrow + c * 64 + hs * 32);
              A.u[1] = *(const v4u*)(inrow + c * 64 + hs * 32 + 16);
            } else {
              A.u[0] = z; A.u[1] = z;
            }
#pragma unroll
            for (int n = 0; n < NT; ++n) {
              // B k order used here (contiguous, matches A): lanes 0-15 K=0..15, lanes 16-31 K=16..31
              const char* wp = bstage + n * (16 * ROWB) + c * 64;
              ABFrag Bf;
              Bf.u[0] = *(const v4u*)(wp);         // ds_load_b128
              Bf.u[1] = *(const v4u*)(wp + 16);    // ds_load_b128
              acc[n] = __builtin_amdgcn_wmma_f32_16x16x32_bf16(
                  false, A.v, false, Bf.v, (short)0, acc[n], false, false);
            }
          }
        }
      }

  // C/D layout: VGPR r -> M = r (lanes 0-15) / r+8 (lanes 16-31), N = lanelo.
  const unsigned ovBase = (unsigned)tile * 16 + hs * 8;
#pragma unroll
  for (int n = 0; n < NT; ++n) {
    const int co = co0 + n * 16 + lanelo;
    const float sc = a.scale[co];
    const float sh = a.shift[co];
#pragma unroll
    for (int r = 0; r < 8; ++r) {
      const unsigned ov = ovBase + r;
      float y = acc[n][r] * sc + sh;
      y = fmaxf(y, 0.f) * a.mask[ov];
      a.out[(size_t)ov * a.CoPad + co] = f2bf(y);
    }
  }
}

// Skip connection: h = y + feat (elementwise over [vox][Cpad] bf16).
__global__ void add_kernel(const bf16_t* __restrict__ xa, const bf16_t* __restrict__ xb,
                           bf16_t* __restrict__ o, long long n) {
  long long i = (long long)blockIdx.x * blockDim.x + threadIdx.x;
  if (i >= n) return;
  o[i] = f2bf(bf2f(xa[i]) + bf2f(xb[i]));
}

// Per-voxel 2-class linear head + softmax(class1) * mask -> f32 output.
__global__ void cls_kernel(const bf16_t* __restrict__ f, const float* __restrict__ wcls,
                           const float* __restrict__ mask, float* __restrict__ out,
                           int C, int CPad, int nvox) {
  int v = blockIdx.x * blockDim.x + threadIdx.x;
  if (v >= nvox) return;
  const bf16_t* row = f + (size_t)v * CPad;
  float l0 = 0.f, l1 = 0.f;
  for (int c = 0; c < C; ++c) {
    float x = bf2f(row[c]);
    l0 += x * wcls[c];
    l1 += x * wcls[C + c];
  }
  float p = 1.f / (1.f + __expf(l0 - l1));
  out[v] = p * mask[v];
}

// f_occu[c][dhw] = h[dhw,t=0,c] + h[dhw,t=1,c]; occu[dhw] = max_c f_occu.
__global__ void final_kernel(const bf16_t* __restrict__ h0, float* __restrict__ out, int n) {
  int i = blockIdx.x * blockDim.x + threadIdx.x;
  if (i >= n) return;
  float mx = -3.4e38f;
#pragma unroll
  for (int c = 0; c < 16; ++c) {
    float s = bf2f(h0[((size_t)i * 2 + 0) * 32 + c]) +
              bf2f(h0[((size_t)i * 2 + 1) * 32 + c]);
    out[(size_t)c * n + i] = s;
    mx = fmaxf(mx, s);
  }
  out[(size_t)16 * n + i] = mx;
}

// ---------------------------------------------------------------------------
// Host orchestration.
// ---------------------------------------------------------------------------
extern "C" void kernel_launch(void* const* d_in, const int* in_sizes, int n_in,
                              void* d_out, int out_size, void* d_ws, size_t ws_size,
                              hipStream_t stream) {
  (void)in_sizes; (void)n_in; (void)out_size; (void)ws_size;
  static const int ENC[5] = {16, 32, 64, 128, 256};
  static const int DIM[5] = {64, 32, 16, 8, 4};
  auto pad32 = [](int c) { return c < 32 ? 32 : c; };
  auto nvox  = [&](int l) { return DIM[l] * DIM[l] * DIM[l] * 2; };
  auto ilog2 = [](int x) { int s = 0; while ((1 << s) < x) ++s; return s; };

  // ---- parse inputs (setup_inputs order, depth-first) ----
  int ii = 0;
  const float* X = (const float*)d_in[ii++];
  const float* W_enc1 = (const float*)d_in[ii++];
  const float* BN1[4];
  for (int k = 0; k < 4; ++k) BN1[k] = (const float*)d_in[ii++];
  const float *W_sp[4], *W_sub[4], *BNa[4][4], *BNb[4][4];
  for (int i = 0; i < 4; ++i) {
    W_sp[i] = (const float*)d_in[ii++];
    for (int k = 0; k < 4; ++k) BNa[i][k] = (const float*)d_in[ii++];
    W_sub[i] = (const float*)d_in[ii++];
    for (int k = 0; k < 4; ++k) BNb[i][k] = (const float*)d_in[ii++];
  }
  const float *DW_inv[4], *DW_sub[4], *DBNa[4][4], *DBNb[4][4];
  for (int j = 0; j < 4; ++j) {
    DW_inv[j] = (const float*)d_in[ii++];
    for (int k = 0; k < 4; ++k) DBNa[j][k] = (const float*)d_in[ii++];
    DW_sub[j] = (const float*)d_in[ii++];
    for (int k = 0; k < 4; ++k) DBNb[j][k] = (const float*)d_in[ii++];
  }
  const float* CLS[4];
  for (int j = 0; j < 4; ++j) CLS[j] = (const float*)d_in[ii++];

  // ---- workspace bump allocator ----
  char* cur = (char*)d_ws;
  auto alloc = [&](size_t bytes) -> char* {
    char* r = cur;
    cur += (bytes + 255) & ~(size_t)255;
    return r;
  };

  // packed weights (bf16, [tap][co][ci_pad])
  bf16_t* PW1 = (bf16_t*)alloc((size_t)81 * 16 * 32 * 2);
  bf16_t *PWsp[4], *PWsub[4], *PDinv[4], *PDsub[4];
  for (int i = 0; i < 4; ++i) {
    PWsp[i]  = (bf16_t*)alloc((size_t)27 * ENC[i + 1] * pad32(ENC[i]) * 2);
    PWsub[i] = (bf16_t*)alloc((size_t)81 * ENC[i + 1] * pad32(ENC[i + 1]) * 2);
  }
  for (int j = 0; j < 4; ++j) {
    PDinv[j] = (bf16_t*)alloc((size_t)27 * ENC[3 - j] * pad32(ENC[4 - j]) * 2);
    PDsub[j] = (bf16_t*)alloc((size_t)81 * ENC[3 - j] * pad32(ENC[3 - j]) * 2);
  }
  // folded BN params
  float *SC1 = (float*)alloc(16 * 4), *SH1 = (float*)alloc(16 * 4);
  float *SCa[4], *SHa[4], *SCb[4], *SHb[4], *DSCa[4], *DSHa[4], *DSCb[4], *DSHb[4];
  for (int i = 0; i < 4; ++i) {
    SCa[i] = (float*)alloc(ENC[i + 1] * 4); SHa[i] = (float*)alloc(ENC[i + 1] * 4);
    SCb[i] = (float*)alloc(ENC[i + 1] * 4); SHb[i] = (float*)alloc(ENC[i + 1] * 4);
  }
  for (int j = 0; j < 4; ++j) {
    DSCa[j] = (float*)alloc(ENC[3 - j] * 4); DSHa[j] = (float*)alloc(ENC[3 - j] * 4);
    DSCb[j] = (float*)alloc(ENC[3 - j] * 4); DSHb[j] = (float*)alloc(ENC[3 - j] * 4);
  }
  // activations / masks
  bf16_t* x32 = (bf16_t*)alloc((size_t)nvox(0) * 32 * 2);
  float* M[5];
  for (int l = 0; l < 5; ++l) M[l] = (float*)alloc((size_t)nvox(l) * 4);
  bf16_t* F[5];
  F[0] = (bf16_t*)alloc((size_t)nvox(0) * 32 * 2);
  for (int l = 1; l < 5; ++l) F[l] = (bf16_t*)alloc((size_t)nvox(l) * pad32(ENC[l]) * 2);
  bf16_t* ESP[4];
  for (int i = 0; i < 4; ++i) ESP[i] = (bf16_t*)alloc((size_t)nvox(i + 1) * pad32(ENC[i + 1]) * 2);
  bf16_t *YA[4], *YB[4], *HH[4];
  for (int j = 0; j < 4; ++j) {
    int lvl = 3 - j;
    size_t bytes = (size_t)nvox(lvl) * pad32(ENC[lvl]) * 2;
    YA[j] = (bf16_t*)alloc(bytes);
    YB[j] = (bf16_t*)alloc(bytes);
    HH[j] = (lvl > 0) ? (bf16_t*)alloc(bytes) : nullptr;
  }

  // ---- launch helpers ----
  auto wpack = [&](const float* w, bf16_t* o, int Co, int Ci, int CiPad, int kt) {
    int per_tap = Co * CiPad;
    dim3 grid((per_tap + 255) / 256, 27, kt);
    wt_pack_kernel<<<grid, 256, 0, stream>>>(w, o, Co, Ci, ilog2(CiPad), kt);
  };
  auto bnprep = [&](const float* const p[4], float* sc, float* sh, int C) {
    bn_prep_kernel<<<(C + 63) / 64, 64, 0, stream>>>(p[0], p[1], p[2], p[3], sc, sh, C);
  };
  auto conv = [&](const bf16_t* in, const bf16_t* wt, bf16_t* out,
                  const float* sc, const float* sh, const float* mask,
                  int lin, int lout, int Ci, int Co, int kt, int stride, int dil) {
    ConvArgs a;
    a.in = in; a.wt = wt; a.out = out; a.scale = sc; a.shift = sh; a.mask = mask;
    a.inDim = DIM[lin];
    a.outDim = DIM[lout];
    a.outSh = ilog2(DIM[lout]);
    a.stride = stride; a.dil = dil;
    a.Co = Co; a.CoPad = pad32(Co);
    int tiles = nvox(lout) / 16;
    a.numTiles = tiles;
    int nc = pad32(Ci) / 32;
    int nt = (Co >= 64) ? 4 : ((Co == 32) ? 2 : 1);
    dim3 grid((tiles + 7) / 8, Co / (16 * nt));
#define DISPATCH(NT, KT, NC)                                              \
    if (nt == NT && kt == KT && nc == NC) {                               \
      conv_wmma_kernel<NT, KT, NC><<<grid, 256, 0, stream>>>(a);          \
      return;                                                             \
    }
    DISPATCH(1, 3, 1) DISPATCH(1, 1, 1)
    DISPATCH(2, 1, 1) DISPATCH(2, 3, 1) DISPATCH(2, 1, 2)
    DISPATCH(4, 1, 1) DISPATCH(4, 1, 2) DISPATCH(4, 3, 2)
    DISPATCH(4, 3, 4) DISPATCH(4, 1, 4) DISPATCH(4, 3, 8) DISPATCH(4, 1, 8)
#undef DISPATCH
  };

  // ---- 1. repack weights + fold BN ----
  wpack(W_enc1, PW1, 16, 9, 32, 3);
  bnprep(BN1, SC1, SH1, 16);
  for (int i = 0; i < 4; ++i) {
    wpack(W_sp[i], PWsp[i], ENC[i + 1], ENC[i], pad32(ENC[i]), 1);
    wpack(W_sub[i], PWsub[i], ENC[i + 1], ENC[i + 1], pad32(ENC[i + 1]), 3);
    bnprep(BNa[i], SCa[i], SHa[i], ENC[i + 1]);
    bnprep(BNb[i], SCb[i], SHb[i], ENC[i + 1]);
  }
  for (int j = 0; j < 4; ++j) {
    wpack(DW_inv[j], PDinv[j], ENC[3 - j], ENC[4 - j], pad32(ENC[4 - j]), 1);
    wpack(DW_sub[j], PDsub[j], ENC[3 - j], ENC[3 - j], pad32(ENC[3 - j]), 3);
    bnprep(DBNa[j], DSCa[j], DSHa[j], ENC[3 - j]);
    bnprep(DBNb[j], DSCb[j], DSHb[j], ENC[3 - j]);
  }

  // ---- 2. pack x, build masks ----
  pack_x_kernel<<<nvox(0) / 256, 256, 0, stream>>>(X, x32, M[0], nvox(0));
  for (int l = 1; l < 5; ++l)
    down_mask_kernel<<<(nvox(l) + 255) / 256, 256, 0, stream>>>(
        M[l - 1], M[l], DIM[l - 1], DIM[l], ilog2(DIM[l]));

  // ---- 3. encoder ----
  conv(x32, PW1, F[0], SC1, SH1, M[0], 0, 0, 9, 16, 3, 1, 1);  // SubM 9->16
  for (int i = 0; i < 4; ++i) {
    conv(F[i], PWsp[i], ESP[i], SCa[i], SHa[i], M[i + 1],
         i, i + 1, ENC[i], ENC[i + 1], 1, 2, 1);               // strided SparseConv
    conv(ESP[i], PWsub[i], F[i + 1], SCb[i], SHb[i], M[i + 1],
         i + 1, i + 1, ENC[i + 1], ENC[i + 1], 3, 1, 1);       // SubMConv
  }

  // ---- 4. decoder + heads ----
  static const size_t probOff[4] = {4456448, 4457472, 4465664, 4531200};
  const bf16_t* h = F[4];
  for (int j = 0; j < 4; ++j) {
    int lvl = 3 - j;
    conv(h, PDinv[j], YA[j], DSCa[j], DSHa[j], M[lvl],
         lvl + 1, lvl, ENC[lvl + 1], ENC[lvl], 1, 1, 2);       // SparseInverseConv (2x up)
    conv(YA[j], PDsub[j], YB[j], DSCb[j], DSHb[j], M[lvl],
         lvl, lvl, ENC[lvl], ENC[lvl], 3, 1, 1);               // SubMConv
    cls_kernel<<<(nvox(lvl) + 255) / 256, 256, 0, stream>>>(
        YB[j], CLS[j], M[lvl], (float*)d_out + probOff[j],
        ENC[lvl], pad32(ENC[lvl]), nvox(lvl));
    if (lvl > 0) {
      long long n = (long long)nvox(lvl) * pad32(ENC[lvl]);
      add_kernel<<<(int)((n + 255) / 256), 256, 0, stream>>>(YB[j], F[lvl], HH[j], n);
      h = HH[j];
    } else {
      h = YB[j];
    }
  }

  // ---- 5. final occupancy outputs ----
  final_kernel<<<(262144 + 255) / 256, 256, 0, stream>>>(YB[3], (float*)d_out, 262144);
}
